// TarsBrain_81166291960259
// MI455X (gfx1250) — matrix operand order repaired and checked by reading, not compiled, not run on hardware
//
#include <hip/hip_runtime.h>
#include <hip/hip_bf16.h>

// ---------------------------------------------------------------------------
// Problem constants (from reference): B=512, S=8, D=1024, L=6, LOOPS=4
// ---------------------------------------------------------------------------
#define D_DIM    1024
#define S_DIM    8
#define B_DIM    512
#define BS_ROWS  (B_DIM * S_DIM)   // 4096 rows of x
#define L_LAYERS 6
#define N_LOOPS  4
#define LN_EPS   1e-5f

typedef __attribute__((ext_vector_type(16))) __bf16 v16bf;
typedef __attribute__((ext_vector_type(8)))  __bf16 v8bf;
typedef __attribute__((ext_vector_type(4)))  __bf16 v4bf;
typedef __attribute__((ext_vector_type(8)))  float  v8f;

// LDS tile row stride in bf16 elements: 40 halves = 80 bytes.
// 80 is a multiple of 16 (keeps b128 LDS accesses 16B-aligned at k-offsets
// 0/16) and 20 dwords/row spreads rows across the 64 LDS banks.
#define LDS_STRIDE 40

// ---------------------------------------------------------------------------
// CDNA5 async copy: per-lane 16B GLOBAL -> LDS, tracked by ASYNCcnt.
//   global_load_async_to_lds_b128  vDstLdsAddr, v[gaddr:gaddr+1], off
// ---------------------------------------------------------------------------
__device__ __forceinline__ void async_copy_b128(unsigned int lds_byte_addr,
                                                const void* gaddr) {
  asm volatile("global_load_async_to_lds_b128 %0, %1, off"
               :
               : "v"(lds_byte_addr), "v"(gaddr)
               : "memory");
}

__device__ __forceinline__ void wait_asynccnt0() {
#if __has_builtin(__builtin_amdgcn_s_wait_asynccnt)
  __builtin_amdgcn_s_wait_asynccnt(0);
#else
  asm volatile("s_wait_asynccnt 0x0" ::: "memory");
#endif
}

// Generic pointer to a __shared__ object carries the DS-space byte offset in
// its low 32 bits on AMDGPU.
__device__ __forceinline__ unsigned int lds_addr_of(const void* p) {
  return (unsigned int)(unsigned long long)p;
}

// ---------------------------------------------------------------------------
// WMMA wrapper: D = A(16x32 bf16) * B(32x16 bf16) + C(16x16 f32)
// ---------------------------------------------------------------------------
__device__ __forceinline__ v8f wmma_bf16(v16bf a, v16bf b, v8f c) {
  return __builtin_amdgcn_wmma_f32_16x16x32_bf16(
      /*neg_a=*/false, a, /*neg_b=*/false, b,
      /*c_mod=*/(short)0, c, /*reuse_a=*/false, /*reuse_b=*/false);
}

// A-fragment (16x32, bf16) from LDS tile stored row-major [m][k]:
//   lanes 0-15: M=lane,    v0..3 = K 0..7,  v4..7 = K 16..23
//   lanes16-31: M=lane-16, v0..3 = K 8..15, v4..7 = K 24..31
__device__ __forceinline__ v16bf frag_a(const __bf16* s, int mbase, int lane) {
  const int m  = mbase + (lane & 15);
  const int kh = (lane >> 4) * 8;
  const __bf16* p = s + m * LDS_STRIDE + kh;
  v8bf lo = *(const v8bf*)(p);
  v8bf hi = *(const v8bf*)(p + 16);
  return __builtin_shufflevector(lo, hi, 0,1,2,3,4,5,6,7,8,9,10,11,12,13,14,15);
}

// B-fragment (32x16, bf16) from LDS tile stored K-major per column [n][k]:
//   lanes 0-15: N=lane,    v0..3 = K 0..7,  v4..7 = K 16..23
//   lanes16-31: N=lane-16, v0..3 = K 8..15, v4..7 = K 24..31
__device__ __forceinline__ v16bf frag_b(const __bf16* s, int nbase, int lane) {
  const int n  = nbase + (lane & 15);
  const int ks = (lane >> 4) * 8;
  const __bf16* p = s + n * LDS_STRIDE + ks;
  v8bf lo = *(const v8bf*)(p);
  v8bf hi = *(const v8bf*)(p + 16);
  return __builtin_shufflevector(lo, hi, 0,1,2,3,4,5,6,7,8,9,10,11,12,13,14,15);
}

// ---------------------------------------------------------------------------
// Kernel 1: fp32 [L][k][n] weight -> bf16 [L][n][k] (transpose + convert)
// grid: (D/32, D/32, L), block: (32, 8)
// ---------------------------------------------------------------------------
__global__ __launch_bounds__(256)
void transpose_cvt_kernel(const float* __restrict__ W,
                          unsigned short* __restrict__ WT_) {
  __bf16* WT = (__bf16*)WT_;
  __shared__ float tile[32][33];
  const int n0 = blockIdx.x * 32;
  const int k0 = blockIdx.y * 32;
  const size_t base = (size_t)blockIdx.z * D_DIM * D_DIM;
#pragma unroll
  for (int i = 0; i < 4; ++i)
    tile[threadIdx.y + 8 * i][threadIdx.x] =
        W[base + (size_t)(k0 + threadIdx.y + 8 * i) * D_DIM + (n0 + threadIdx.x)];
  __syncthreads();
#pragma unroll
  for (int i = 0; i < 4; ++i)
    WT[base + (size_t)(n0 + threadIdx.y + 8 * i) * D_DIM + (k0 + threadIdx.x)] =
        (__bf16)tile[threadIdx.x][threadIdx.y + 8 * i];
}

// ---------------------------------------------------------------------------
// Kernel 2: LayerNorm over D, output bf16 xn. One 256-thread block per row.
// ---------------------------------------------------------------------------
__global__ __launch_bounds__(256)
void layernorm_bf16_kernel(const float* __restrict__ x,
                           const float* __restrict__ gamma,
                           const float* __restrict__ beta,
                           unsigned short* __restrict__ xn_) {
  __bf16* xn = (__bf16*)xn_;
  const int row = blockIdx.x;
  const int tid = threadIdx.x;
  const size_t off = (size_t)row * D_DIM + tid * 4;
  float4 v = *(const float4*)&x[off];
  float s = v.x + v.y + v.z + v.w;
  float q = v.x * v.x + v.y * v.y + v.z * v.z + v.w * v.w;

  __shared__ float rs[256];
  __shared__ float rq[256];
  rs[tid] = s; rq[tid] = q;
  __syncthreads();
  for (int o = 128; o > 0; o >>= 1) {
    if (tid < o) { rs[tid] += rs[tid + o]; rq[tid] += rq[tid + o]; }
    __syncthreads();
  }
  const float mu   = rs[0] * (1.0f / D_DIM);
  const float var  = rq[0] * (1.0f / D_DIM) - mu * mu;
  const float rstd = rsqrtf(var + LN_EPS);

  float4 gv = *(const float4*)&gamma[tid * 4];
  float4 bv = *(const float4*)&beta[tid * 4];
  v4bf o;
  o[0] = (__bf16)((v.x - mu) * rstd * gv.x + bv.x);
  o[1] = (__bf16)((v.y - mu) * rstd * gv.y + bv.y);
  o[2] = (__bf16)((v.z - mu) * rstd * gv.z + bv.z);
  o[3] = (__bf16)((v.w - mu) * rstd * gv.w + bv.w);
  *(v4bf*)&xn[off] = o;
}

// ---------------------------------------------------------------------------
// Kernel 3: fused dual GEMM + gate.
//   G = xn @ Wg + bg ; T = xn @ Wt + bt
//   x = x * (1 - sigmoid(G)) + tanh(T) * sigmoid(G)   (in place)
// Block tile 128x128, 8 waves, wave tile 32x64.
// K staged 32 at a time into DOUBLE-BUFFERED LDS via
// global_load_async_to_lds_b128 (ASYNCcnt): iteration k's copies overlap
// iteration k-1's WMMAs; one barrier per K-step.
// grid: (BS/128, D/128), block: 256
// ---------------------------------------------------------------------------
__global__ __launch_bounds__(256)
void gemm_gate_kernel(const unsigned short* __restrict__ xn_,
                      const unsigned short* __restrict__ wgT_,
                      const unsigned short* __restrict__ wtT_,
                      const float* __restrict__ bg,
                      const float* __restrict__ bt,
                      float* __restrict__ x) {
  const __bf16* xn  = (const __bf16*)xn_;
  const __bf16* wgT = (const __bf16*)wgT_;
  const __bf16* wtT = (const __bf16*)wtT_;

  __shared__ __bf16 As [2][128 * LDS_STRIDE];   // activations [m][k]
  __shared__ __bf16 Bgs[2][128 * LDS_STRIDE];   // Wg tile     [n][k]
  __shared__ __bf16 Bts[2][128 * LDS_STRIDE];   // Wt tile     [n][k]

  const int tid  = threadIdx.x;
  const int lane = tid & 31;
  const int wid  = tid >> 5;
  const int m0 = blockIdx.x * 128;
  const int n0 = blockIdx.y * 128;
  const int mw = (wid & 3) * 32;   // wave M offset in block tile
  const int nw = (wid >> 2) * 64;  // wave N offset in block tile

  // --- staging coordinates: 512 x 16B chunks per tile, 2 per thread -------
  const int r0  = tid >> 2;                 // chunk tid
  const int ko0 = (tid & 3) * 8;
  const int r1  = (tid + 256) >> 2;         // chunk tid+256
  const int ko1 = ((tid + 256) & 3) * 8;

  // global source pointers (advance by 32 bf16 per K-step)
  const __bf16* gA0 = &xn [(size_t)(m0 + r0) * D_DIM + ko0];
  const __bf16* gA1 = &xn [(size_t)(m0 + r1) * D_DIM + ko1];
  const __bf16* gG0 = &wgT[(size_t)(n0 + r0) * D_DIM + ko0];
  const __bf16* gG1 = &wgT[(size_t)(n0 + r1) * D_DIM + ko1];
  const __bf16* gT0 = &wtT[(size_t)(n0 + r0) * D_DIM + ko0];
  const __bf16* gT1 = &wtT[(size_t)(n0 + r1) * D_DIM + ko1];

  // LDS destination byte addresses per buffer (constant across K-steps)
  unsigned int dA0[2], dA1[2], dG0[2], dG1[2], dT0[2], dT1[2];
#pragma unroll
  for (int b = 0; b < 2; ++b) {
    dA0[b] = lds_addr_of(&As [b][r0 * LDS_STRIDE + ko0]);
    dA1[b] = lds_addr_of(&As [b][r1 * LDS_STRIDE + ko1]);
    dG0[b] = lds_addr_of(&Bgs[b][r0 * LDS_STRIDE + ko0]);
    dG1[b] = lds_addr_of(&Bgs[b][r1 * LDS_STRIDE + ko1]);
    dT0[b] = lds_addr_of(&Bts[b][r0 * LDS_STRIDE + ko0]);
    dT1[b] = lds_addr_of(&Bts[b][r1 * LDS_STRIDE + ko1]);
  }

#define ISSUE_TILE_COPIES(buf)                                           \
  do {                                                                   \
    async_copy_b128(dA0[buf], gA0);  async_copy_b128(dA1[buf], gA1);     \
    async_copy_b128(dG0[buf], gG0);  async_copy_b128(dG1[buf], gG1);     \
    async_copy_b128(dT0[buf], gT0);  async_copy_b128(dT1[buf], gT1);     \
    gA0 += 32; gA1 += 32; gG0 += 32;                                     \
    gG1 += 32; gT0 += 32; gT1 += 32;                                     \
  } while (0)

  v8f accg[2][4], acct[2][4];
  const v8f vzero = {0.f, 0.f, 0.f, 0.f, 0.f, 0.f, 0.f, 0.f};
#pragma unroll
  for (int mt = 0; mt < 2; ++mt)
#pragma unroll
    for (int nt = 0; nt < 4; ++nt) { accg[mt][nt] = vzero; acct[mt][nt] = vzero; }

  // Prologue: stage K-tile 0 into buffer 0.
  ISSUE_TILE_COPIES(0);

  int cur = 0;
  for (int k0 = 0; k0 < D_DIM; k0 += 32, cur ^= 1) {
    wait_asynccnt0();        // my copies into buf[cur] have landed
    __syncthreads();         // everyone's copies landed; prev reads of
                             // buf[cur^1] are complete -> safe to overwrite
    if (k0 + 32 < D_DIM) {
      ISSUE_TILE_COPIES(cur ^ 1);          // overlaps with WMMAs below
      __builtin_prefetch(gG0, 0, 1);       // warm K-tile k0+64 toward L2
      __builtin_prefetch(gT0, 0, 1);
    }

    const __bf16* Asb = As [cur];
    const __bf16* Bgb = Bgs[cur];
    const __bf16* Btb = Bts[cur];
    v16bf a0 = frag_a(Asb, mw,      lane);
    v16bf a1 = frag_a(Asb, mw + 16, lane);
#pragma unroll
    for (int nt = 0; nt < 4; ++nt) {
      v16bf bG = frag_b(Bgb, nw + nt * 16, lane);
      v16bf bT = frag_b(Btb, nw + nt * 16, lane);
      accg[0][nt] = wmma_bf16(a0, bG, accg[0][nt]);
      accg[1][nt] = wmma_bf16(a1, bG, accg[1][nt]);
      acct[0][nt] = wmma_bf16(a0, bT, acct[0][nt]);
      acct[1][nt] = wmma_bf16(a1, bT, acct[1][nt]);
    }
  }
#undef ISSUE_TILE_COPIES

  // Epilogue: C/D layout -> lanes 0-15 hold M=j, lanes 16-31 hold M=8+j.
  const int rbase = m0 + mw + ((lane >> 4) * 8);
  const int cbase = n0 + nw + (lane & 15);
#pragma unroll
  for (int mt = 0; mt < 2; ++mt) {
#pragma unroll
    for (int nt = 0; nt < 4; ++nt) {
      const int col = cbase + nt * 16;
      const float bgv = bg[col];
      const float btv = bt[col];
#pragma unroll
      for (int j = 0; j < 8; ++j) {
        const int row = rbase + mt * 16 + j;
        const float g = 1.0f / (1.0f + __expf(-(accg[mt][nt][j] + bgv)));
        const float t = tanhf(acct[mt][nt][j] + btv);
        float* p = &x[(size_t)row * D_DIM + col];
        *p = *p * (1.0f - g) + t * g;
      }
    }
  }
}

// ---------------------------------------------------------------------------
// Kernel 4: confidence head: sigmoid( mean_s(x) @ Wc + bc ) -> out[b]
// One 256-thread block per batch element.
// ---------------------------------------------------------------------------
__global__ __launch_bounds__(256)
void conf_kernel(const float* __restrict__ x, const float* __restrict__ Wc,
                 const float* __restrict__ bc, float* __restrict__ out) {
  const int b   = blockIdx.x;
  const int tid = threadIdx.x;
  float acc = 0.f;
  for (int d = tid; d < D_DIM; d += 256) {
    float m = 0.f;
#pragma unroll
    for (int s = 0; s < S_DIM; ++s)
      m += x[(size_t)(b * S_DIM + s) * D_DIM + d];
    acc += (m * (1.0f / S_DIM)) * Wc[d];
  }
  __shared__ float red[256];
  red[tid] = acc;
  __syncthreads();
  for (int o = 128; o > 0; o >>= 1) {
    if (tid < o) red[tid] += red[tid + o];
    __syncthreads();
  }
  if (tid == 0) out[b] = 1.0f / (1.0f + __expf(-(red[0] + bc[0])));
}

// ---------------------------------------------------------------------------
// Host driver
// Inputs: x, gamma, beta, Wg, bg, Wt, bt, Wc, bc  (all fp32)
// Output: [ x_final (B*S*D) | confs (LOOPS*B) ]
// Workspace: xn_bf16 (8MB) | WgT_bf16 (12MB) | WtT_bf16 (12MB)
// ---------------------------------------------------------------------------
extern "C" void kernel_launch(void* const* d_in, const int* in_sizes, int n_in,
                              void* d_out, int out_size, void* d_ws, size_t ws_size,
                              hipStream_t stream) {
  const float* x_in  = (const float*)d_in[0];
  const float* gamma = (const float*)d_in[1];
  const float* beta  = (const float*)d_in[2];
  const float* Wg    = (const float*)d_in[3];
  const float* bg    = (const float*)d_in[4];
  const float* Wt    = (const float*)d_in[5];
  const float* bt    = (const float*)d_in[6];
  const float* Wc    = (const float*)d_in[7];
  const float* bc    = (const float*)d_in[8];

  float* x     = (float*)d_out;                       // x updated in place here
  float* confs = x + (size_t)BS_ROWS * D_DIM;         // LOOPS*B tail

  char* ws = (char*)d_ws;
  unsigned short* xn  = (unsigned short*)ws;                                   // 8 MB
  unsigned short* WgT = (unsigned short*)(ws + (size_t)8  * 1024 * 1024);      // 12 MB
  unsigned short* WtT = (unsigned short*)(ws + (size_t)20 * 1024 * 1024);      // 12 MB

  // x_work = input x (never mutate d_in)
  hipMemcpyAsync(x, x_in, (size_t)BS_ROWS * D_DIM * sizeof(float),
                 hipMemcpyDeviceToDevice, stream);

  // One-time weight transpose+convert to bf16 [l][n][k]
  {
    dim3 tg(D_DIM / 32, D_DIM / 32, L_LAYERS);
    dim3 tb(32, 8);
    transpose_cvt_kernel<<<tg, tb, 0, stream>>>(Wg, WgT);
    transpose_cvt_kernel<<<tg, tb, 0, stream>>>(Wt, WtT);
  }

  const dim3 ggrid(BS_ROWS / 128, D_DIM / 128);
  for (int loop = 0; loop < N_LOOPS; ++loop) {
    for (int l = 0; l < L_LAYERS; ++l) {
      layernorm_bf16_kernel<<<BS_ROWS, 256, 0, stream>>>(
          x, gamma + (size_t)l * D_DIM, beta + (size_t)l * D_DIM, xn);
      gemm_gate_kernel<<<ggrid, 256, 0, stream>>>(
          xn,
          WgT + (size_t)l * D_DIM * D_DIM,
          WtT + (size_t)l * D_DIM * D_DIM,
          bg + (size_t)l * D_DIM, bt + (size_t)l * D_DIM, x);
    }
    conf_kernel<<<B_DIM, 256, 0, stream>>>(x, Wc, bc, confs + (size_t)loop * B_DIM);
  }
}
